// LTCODE_79877801771610
// MI455X (gfx1250) — matrix operand order repaired and checked by reading, not compiled, not run on hardware
//
#include <hip/hip_runtime.h>

#define BATCH  8192
#define HIDDEN 1024
#define INPUT  1024
#define LN_EPS 1e-5f

typedef __attribute__((ext_vector_type(16))) __bf16 v16bf;
typedef __attribute__((ext_vector_type(8)))  float  v8f;

// ---------------------------------------------------------------------------
// CDNA5 hardware tanh (v_tanh_f32, new in gfx1250) with libm fallback.
// sigmoid(x) = 0.5 * (1 + tanh(0.5 x))  -> 1 TRANS op + 2 FMA instead of
// the exp/rcp chain; TRANS ops co-execute with WMMA on CDNA5.
// ---------------------------------------------------------------------------
#if __has_builtin(__builtin_amdgcn_tanhf)
__device__ __forceinline__ float fast_tanh(float x) { return __builtin_amdgcn_tanhf(x); }
#else
__device__ __forceinline__ float fast_tanh(float x) { return tanhf(x); }
#endif
__device__ __forceinline__ float fast_sigmoid(float x) {
    return fmaf(0.5f, fast_tanh(0.5f * x), 0.5f);
}

// ---------------------------------------------------------------------------
// fp32 -> bf16 with round-to-nearest-even
// ---------------------------------------------------------------------------
__device__ __forceinline__ unsigned short f32_to_bf16_rne(float f) {
    unsigned int u = __float_as_uint(f);
    u += 0x7FFFu + ((u >> 16) & 1u);
    return (unsigned short)(u >> 16);
}

__global__ void LTC_cvt_bf16_kernel(const float* __restrict__ src,
                                    unsigned short* __restrict__ dst, int n4) {
    int i = blockIdx.x * blockDim.x + threadIdx.x;
    if (i >= n4) return;
    float4 f = reinterpret_cast<const float4*>(src)[i];
    ushort4 o;
    o.x = f32_to_bf16_rne(f.x);
    o.y = f32_to_bf16_rne(f.y);
    o.z = f32_to_bf16_rne(f.z);
    o.w = f32_to_bf16_rne(f.w);
    reinterpret_cast<ushort4*>(dst)[i] = o;
}

// ---------------------------------------------------------------------------
// WMMA fragment loaders (bf16, 16x16x32 shape), per CDNA5 ISA
// A 16x32: lanes 0-15 -> row=lane, K {0..7,16..23}; lanes 16-31 -> K {8..15,24..31}
// B 32x16: lanes 0-15 -> col=lane, K {0..7,16..23}; lanes 16-31 -> K {8..15,24..31}
// ---------------------------------------------------------------------------
__device__ __forceinline__ v16bf load_a_frag(const unsigned short* p) {
    union { uint4 u[2]; v16bf v; } f;
    f.u[0] = *reinterpret_cast<const uint4*>(p);       // K+0 .. K+7
    f.u[1] = *reinterpret_cast<const uint4*>(p + 16);  // K+16 .. K+23
    return f.v;
}
__device__ __forceinline__ v16bf load_b_frag(const unsigned short* p) {
    union { uint4 u[2]; v16bf v; } f;
    f.u[0] = *reinterpret_cast<const uint4*>(p);       // K+0 .. K+7
    f.u[1] = *reinterpret_cast<const uint4*>(p + 16);  // K+16 .. K+23
    return f.v;
}

#define WMMA_BF16(a, b, c) \
    __builtin_amdgcn_wmma_f32_16x16x32_bf16(false, (a), false, (b), (short)0, (c), false, false)

// ---------------------------------------------------------------------------
// Fused LTC cell: 3 GEMMs (bf16 WMMA, f32 acc) + gate/tanh/tau + LayerNorm
// Workgroup: 256 threads (8 wave32), tile = 32 rows x 1024 cols.
// Wave w owns columns [w*128, w*128+128), split into 8 sub-tiles of 16.
// Dynamic LDS: 32*1024 floats (128 KB) for dh staging before LayerNorm.
// ---------------------------------------------------------------------------
__global__ __launch_bounds__(256, 1)
void LTC_fused_wmma_kernel(const float* __restrict__ h32,
                           const float* __restrict__ log_tau,
                           const float* __restrict__ gamma,
                           const float* __restrict__ beta,
                           const unsigned short* __restrict__ hb,
                           const unsigned short* __restrict__ xb,
                           const unsigned short* __restrict__ wrec,
                           const unsigned short* __restrict__ win,
                           const unsigned short* __restrict__ wgate,
                           float* __restrict__ out) {
    extern __shared__ float s_dh[];                 // [32][HIDDEN]

    const int lane = threadIdx.x & 31;
    const int wave = threadIdx.x >> 5;              // 0..7
    const int r0   = blockIdx.x * 32;               // batch-row base

    // A-fragment lane mapping
    const int a_row  = lane & 15;
    const int a_koff = (lane >> 4) << 3;            // 0 or 8
    // B-fragment lane mapping
    const int b_col  = lane & 15;
    const int b_koff = (lane >> 4) << 3;            // 0 or 8

    const unsigned short* hA0 = hb + (size_t)(r0 +      a_row) * HIDDEN + a_koff;
    const unsigned short* hA1 = hb + (size_t)(r0 + 16 + a_row) * HIDDEN + a_koff;
    const unsigned short* xA0 = xb + (size_t)(r0 +      a_row) * INPUT  + a_koff;
    const unsigned short* xA1 = xb + (size_t)(r0 + 16 + a_row) * INPUT  + a_koff;

    for (int nt = 0; nt < 8; ++nt) {
        const int col_base = wave * 128 + nt * 16;
        const unsigned short* pWr = wrec  + (size_t)(col_base + b_col) * HIDDEN + b_koff;
        const unsigned short* pWi = win   + (size_t)(col_base + b_col) * INPUT  + b_koff;
        const unsigned short* pWg = wgate + (size_t)(col_base + b_col) * HIDDEN + b_koff;

        v8f p0 = {}, p1 = {}, g0 = {}, g1 = {};

        for (int k0 = 0; k0 < HIDDEN; k0 += 32) {
            // prefetch weight stream a few K-steps ahead (global_prefetch_b8)
            __builtin_prefetch(pWr + k0 + 256, 0, 1);
            __builtin_prefetch(pWi + k0 + 256, 0, 1);
            __builtin_prefetch(pWg + k0 + 256, 0, 1);

            v16bf br  = load_b_frag(pWr + k0);
            v16bf bi  = load_b_frag(pWi + k0);
            v16bf bg  = load_b_frag(pWg + k0);
            v16bf ah0 = load_a_frag(hA0 + k0);
            v16bf ah1 = load_a_frag(hA1 + k0);
            v16bf ax0 = load_a_frag(xA0 + k0);
            v16bf ax1 = load_a_frag(xA1 + k0);

            p0 = WMMA_BF16(ah0, br, p0);            // pre += h  @ W_rec^T
            p1 = WMMA_BF16(ah1, br, p1);
            p0 = WMMA_BF16(ax0, bi, p0);            // pre += x  @ W_in^T
            p1 = WMMA_BF16(ax1, bi, p1);
            g0 = WMMA_BF16(ah0, bg, g0);            // gate = h @ W_gate^T
            g1 = WMMA_BF16(ah1, bg, g1);
        }

        // ---- fused epilogue for this 16-column sub-tile ----
        // C layout: VGPR i, lanes 0-15 -> row i, col lane; lanes 16-31 -> row i+8
        const int   col   = col_base + (lane & 15);
        const int   rhalf = (lane >> 4) << 3;       // 0 or 8
        const float itau  = __expf(-log_tau[col]);

        #pragma unroll
        for (int i = 0; i < 8; ++i) {
            const int lr0 = rhalf + i;              // local row of tile 0
            const int lr1 = 16 + rhalf + i;         // local row of tile 1
            float hv0 = h32[(size_t)(r0 + lr0) * HIDDEN + col];
            float hv1 = h32[(size_t)(r0 + lr1) * HIDDEN + col];
            float gv0 = fast_sigmoid(g0[i]);
            float gv1 = fast_sigmoid(g1[i]);
            s_dh[lr0 * HIDDEN + col] = (gv0 * fast_tanh(p0[i]) - hv0) * itau;
            s_dh[lr1 * HIDDEN + col] = (gv1 * fast_tanh(p1[i]) - hv1) * itau;
        }
    }

    __syncthreads();

    // ---- LayerNorm over HIDDEN per row: 8 waves x 4 rows, wave32 reduction ----
    for (int r = wave * 4; r < wave * 4 + 4; ++r) {
        float s = 0.0f, sq = 0.0f;
        for (int j = lane; j < HIDDEN; j += 32) {
            float v = s_dh[r * HIDDEN + j];
            s  += v;
            sq += v * v;
        }
        #pragma unroll
        for (int m = 16; m >= 1; m >>= 1) {
            s  += __shfl_xor(s,  m, 32);
            sq += __shfl_xor(sq, m, 32);
        }
        const float mu   = s * (1.0f / HIDDEN);
        const float var  = sq * (1.0f / HIDDEN) - mu * mu;
        const float rstd = rsqrtf(var + LN_EPS);
        float* op = out + (size_t)(r0 + r) * HIDDEN;
        for (int j = lane; j < HIDDEN; j += 32) {
            float v = s_dh[r * HIDDEN + j];
            op[j] = (v - mu) * rstd * gamma[j] + beta[j];
        }
    }
}

// ---------------------------------------------------------------------------
extern "C" void kernel_launch(void* const* d_in, const int* in_sizes, int n_in,
                              void* d_out, int out_size, void* d_ws, size_t ws_size,
                              hipStream_t stream) {
    (void)in_sizes; (void)n_in; (void)out_size; (void)ws_size;

    const float* h       = (const float*)d_in[0];
    const float* x       = (const float*)d_in[1];
    const float* W_in    = (const float*)d_in[2];
    const float* W_rec   = (const float*)d_in[3];
    const float* W_gate  = (const float*)d_in[4];
    const float* log_tau = (const float*)d_in[5];
    const float* gamma   = (const float*)d_in[6];
    const float* beta    = (const float*)d_in[7];
    float*       out     = (float*)d_out;

    // Workspace layout (bf16 copies): h, x, W_rec, W_in, W_gate  (~38 MB)
    unsigned short* hb  = (unsigned short*)d_ws;
    unsigned short* xb  = hb  + (size_t)BATCH  * HIDDEN;
    unsigned short* wrb = xb  + (size_t)BATCH  * INPUT;
    unsigned short* wib = wrb + (size_t)HIDDEN * HIDDEN;
    unsigned short* wgb = wib + (size_t)HIDDEN * INPUT;

    auto cvt = [&](const float* s, unsigned short* d, size_t n) {
        int n4 = (int)(n / 4);
        LTC_cvt_bf16_kernel<<<(n4 + 255) / 256, 256, 0, stream>>>(s, d, n4);
    };
    cvt(h,      hb,  (size_t)BATCH  * HIDDEN);
    cvt(x,      xb,  (size_t)BATCH  * INPUT);
    cvt(W_rec,  wrb, (size_t)HIDDEN * HIDDEN);
    cvt(W_in,   wib, (size_t)HIDDEN * INPUT);
    cvt(W_gate, wgb, (size_t)HIDDEN * HIDDEN);

    const size_t lds_bytes = 32u * HIDDEN * sizeof(float);   // 128 KB / WG
    LTC_fused_wmma_kernel<<<BATCH / 32, 256, lds_bytes, stream>>>(
        h, log_tau, gamma, beta, hb, xb, wrb, wib, wgb, out);
}
